// simple_GAT_25366076850193
// MI455X (gfx1250) — matrix operand order repaired and checked by reading, not compiled, not run on hardware
//
#include <hip/hip_runtime.h>
#include <hip/hip_bf16.h>
#include <math.h>

// ---------------------------------------------------------------------------
// Problem constants (from reference)
// ---------------------------------------------------------------------------
#define NN   50000
#define EE   800000
#define INF_ 128
#define HH   64
#define EL   (EE + NN)          // edges + self loops
#define BN_EPS 1e-5f

typedef __attribute__((ext_vector_type(16))) __bf16 v16bf;
typedef __attribute__((ext_vector_type(8)))  __bf16 v8bf;
typedef __attribute__((ext_vector_type(8)))  float  v8f;

// ---------------------------------------------------------------------------
// WMMA GEMM: C[M,64] = A[M,K] @ B[K,64]  (row-major f32, bf16 matrix cores)
// K is a compile-time constant (128 or 64) -> K-loop fully unrolled: the wave
// issues K/32 back-to-back v_wmma_f32_16x16x32_bf16 with A-loads pipelined.
// Block = 128 threads = 4 waves; block computes a 16x64 tile of C.
//
// B panel is converted to bf16 ONCE per block into LDS, stored TRANSPOSED
// with a padded stride (LDB = K+8):
//   - lane's B fragment (col fixed, k = kb..kb+15) = 32 contiguous bytes,
//     16B-aligned  ->  two ds_load_b128 per K-step (vs 16x ds_load_u16)
//   - column stride (K+8)*2 bytes is a multiple of 16, and spreads the 16
//     active columns across 16 distinct LDS banks.
// ---------------------------------------------------------------------------
template <int K>
__global__ __launch_bounds__(128) void gemm_wmma_bf16(
    const float* __restrict__ A, const float* __restrict__ B,
    float* __restrict__ C, int M)
{
    constexpr int LDB = K + 8;                       // padded stride (elements)
    __shared__ alignas(16) __bf16 Bs[64 * LDB];      // transposed bf16 panel
    const int tid = threadIdx.x;

    #pragma unroll
    for (int i = tid; i < K * 64; i += 128) {        // fill: Bs[n][k] = B[k][n]
        const int k = i >> 6;
        const int n = i & 63;
        Bs[n * LDB + k] = (__bf16)B[(size_t)k * 64 + n];
    }
    __syncthreads();

    const int wave = tid >> 5;
    const int lane = tid & 31;
    const int row0 = blockIdx.x * 16;
    const int n0   = wave * 16;                 // wave's 16-col slice of the 64
    const int m    = lane & 15;
    const int half = lane >> 4;                 // 0 or 1

    const float*  arow = A + (size_t)(row0 + m) * K;
    const int     col  = n0 + m;
    const __bf16* bcol = Bs + col * LDB;

    v8f acc = {};
    #pragma unroll
    for (int k0 = 0; k0 < K; k0 += 32) {
        // ---- A fragment (16x32 bf16): lane(row m) holds K = ka..ka+15
        v16bf a;
        const int ka = k0 + half * 16;
        #pragma unroll
        for (int i = 0; i < 8; ++i) a[i]     = (__bf16)arow[ka + i];
        #pragma unroll
        for (int i = 0; i < 8; ++i) a[8 + i] = (__bf16)arow[ka + 8 + i];

        // ---- B fragment (32x16 bf16): lane(col) holds K = kb..kb+15
        //      two 16B vector loads from the transposed LDS panel
        const int kb = k0 + half * 16;
        const v8bf b0 = *(const v8bf*)(bcol + kb);
        const v8bf b1 = *(const v8bf*)(bcol + kb + 8);
        v16bf b;
        #pragma unroll
        for (int i = 0; i < 8; ++i) { b[i] = b0[i]; b[8 + i] = b1[i]; }

        acc = __builtin_amdgcn_wmma_f32_16x16x32_bf16(
            /*neg_a=*/false, a, /*neg_b=*/false, b,
            /*c_mod=*/(short)0, acc, /*reuse_a=*/false, /*reuse_b=*/false);
    }

    // ---- store D: lane holds C[row0 + half*8 + j][n0 + m], j = 0..7
    float* crow = C + (size_t)(row0 + half * 8) * 64 + col;
    #pragma unroll
    for (int j = 0; j < 8; ++j) crow[(size_t)j * 64] = acc[j];
}

// ---------------------------------------------------------------------------
// Per-node attention dot products: asn = h @ a_s, adn = h @ a_d
// ---------------------------------------------------------------------------
__global__ __launch_bounds__(256) void attn_dots(
    const float* __restrict__ h, const float* __restrict__ a_s,
    const float* __restrict__ a_d, float* __restrict__ asn,
    float* __restrict__ adn, int n)
{
    int i = blockIdx.x * blockDim.x + threadIdx.x;
    if (i >= n) return;
    const float* hr = h + (size_t)i * 64;
    float s0 = 0.f, s1 = 0.f;
    #pragma unroll
    for (int f = 0; f < 64; ++f) {
        float hv = hr[f];
        s0 += hv * a_s[f];
        s1 += hv * a_d[f];
    }
    asn[i] = s0;
    adn[i] = s1;
}

// ---------------------------------------------------------------------------
// Per-layer accumulator init: agg = 0, emax = -inf, denom = 0
// ---------------------------------------------------------------------------
__global__ __launch_bounds__(256) void init_layer(
    float* __restrict__ agg, float* __restrict__ emax,
    float* __restrict__ denom, int n)
{
    int idx = blockIdx.x * blockDim.x + threadIdx.x;
    if (idx < n * 64) agg[idx] = 0.f;
    if (idx < n) { emax[idx] = -3.0e38f; denom[idx] = 0.f; }
}

__device__ __forceinline__ void edge_sd(const int* __restrict__ src,
                                        const int* __restrict__ dst,
                                        int e, int& s, int& d)
{
    if (e < EE) { s = src[e]; d = dst[e]; }
    else        { s = d = e - EE; }         // self loop
}

__device__ __forceinline__ float leaky02(float v)
{
    return v > 0.f ? v : 0.2f * v;
}

// ---------------------------------------------------------------------------
// Edge pass 1: segment max of leaky_relu(asn[src] + adn[dst]) over dst
// ---------------------------------------------------------------------------
__global__ __launch_bounds__(256) void edge_max(
    const int* __restrict__ src, const int* __restrict__ dst,
    const float* __restrict__ asn, const float* __restrict__ adn,
    float* __restrict__ emax)
{
    int e = blockIdx.x * blockDim.x + threadIdx.x;
    if (e >= EL) return;
    int s, d; edge_sd(src, dst, e, s, d);
    float v = leaky02(asn[s] + adn[d]);
    atomicMax(emax + d, v);
}

// ---------------------------------------------------------------------------
// Edge pass 2: denom[d] += exp(v - emax[d])
// ---------------------------------------------------------------------------
__global__ __launch_bounds__(256) void edge_denom(
    const int* __restrict__ src, const int* __restrict__ dst,
    const float* __restrict__ asn, const float* __restrict__ adn,
    const float* __restrict__ emax, float* __restrict__ denom)
{
    int e = blockIdx.x * blockDim.x + threadIdx.x;
    if (e >= EL) return;
    int s, d; edge_sd(src, dst, e, s, d);
    float v = leaky02(asn[s] + adn[d]);
    atomicAdd(denom + d, __expf(v - emax[d]));
}

// ---------------------------------------------------------------------------
// Edge pass 3: one wave per edge; agg[d] += alpha * h[s]  (64 features,
// lanes do features f and f+32 -> coalesced global_atomic_add_f32)
// ---------------------------------------------------------------------------
__global__ __launch_bounds__(256) void edge_aggregate(
    const int* __restrict__ src, const int* __restrict__ dst,
    const float* __restrict__ asn, const float* __restrict__ adn,
    const float* __restrict__ emax, const float* __restrict__ denom,
    const float* __restrict__ h, float* __restrict__ agg)
{
    int e    = (blockIdx.x * blockDim.x + threadIdx.x) >> 5;
    int lane = threadIdx.x & 31;
    if (e >= EL) return;
    int s, d; edge_sd(src, dst, e, s, d);
    float v     = leaky02(asn[s] + adn[d]);
    float ex    = __expf(v - emax[d]);
    float alpha = ex / (denom[d] + 1e-16f);
    const float* hs = h + (size_t)s * 64;
    float*       ag = agg + (size_t)d * 64;
    atomicAdd(ag + lane,      alpha * hs[lane]);
    atomicAdd(ag + lane + 32, alpha * hs[lane + 32]);
}

// ---------------------------------------------------------------------------
// Layer epilogue: xp = bn(relu(agg + bias)) [+ proj + projb]
// ---------------------------------------------------------------------------
__global__ __launch_bounds__(256) void finish_layer(
    const float* __restrict__ agg, const float* __restrict__ bias,
    const float* __restrict__ g, const float* __restrict__ b,
    const float* __restrict__ mu, const float* __restrict__ var,
    const float* __restrict__ proj, const float* __restrict__ projb,
    float* __restrict__ xp, int useProj, int n)
{
    int idx = blockIdx.x * blockDim.x + threadIdx.x;
    if (idx >= n * 64) return;
    int f = idx & 63;
    float v = agg[idx] + bias[f];
    v = fmaxf(v, 0.f);
    v = (v - mu[f]) * rsqrtf(var[f] + BN_EPS) * g[f] + b[f];
    if (useProj) v += proj[idx] + projb[f];
    xp[idx] = v;
}

// ---------------------------------------------------------------------------
// Global mean pool (stage 1): partial sums -> atomicAdd into pooled[64]
// ---------------------------------------------------------------------------
__global__ __launch_bounds__(64) void init_pooled(float* __restrict__ pooled)
{
    if (threadIdx.x < 64) pooled[threadIdx.x] = 0.f;
}

__global__ __launch_bounds__(256) void pool_mean(
    const float* __restrict__ xp, float* __restrict__ pooled, int n)
{
    __shared__ float smem[256];
    int tid = threadIdx.x;
    int f   = tid & 63;
    int r0  = tid >> 6;                       // 0..3
    int base = blockIdx.x * 1024;
    float s = 0.f;
    for (int r = r0; r < 1024; r += 4) {
        int row = base + r;
        if (row < n) s += xp[(size_t)row * 64 + f];
    }
    smem[tid] = s;
    __syncthreads();
    if (tid < 64)
        atomicAdd(pooled + f, smem[tid] + smem[tid + 64] +
                              smem[tid + 128] + smem[tid + 192]);
}

// ---------------------------------------------------------------------------
// MLP head (1 wave): g = pooled/N; h1 = bn(relu(g@hW1+hb1)); out = h1@hW2+hb2
// ---------------------------------------------------------------------------
__global__ __launch_bounds__(32) void head_kernel(
    const float* __restrict__ pooled,
    const float* __restrict__ hW1, const float* __restrict__ hb1,
    const float* __restrict__ g,   const float* __restrict__ b,
    const float* __restrict__ mu,  const float* __restrict__ var,
    const float* __restrict__ hW2, const float* __restrict__ hb2,
    float* __restrict__ out, int n)
{
    __shared__ float h1s[32];
    int j = threadIdx.x;
    float acc = hb1[j];
    const float inv_n = 1.0f / (float)n;
    #pragma unroll
    for (int k = 0; k < 64; ++k)
        acc += (pooled[k] * inv_n) * hW1[k * 32 + j];
    acc = fmaxf(acc, 0.f);
    acc = (acc - mu[j]) * rsqrtf(var[j] + BN_EPS) * g[j] + b[j];
    h1s[j] = acc * hW2[j];
    __syncthreads();
    if (j == 0) {
        float s = hb2[0];
        #pragma unroll
        for (int k = 0; k < 32; ++k) s += h1s[k];
        out[0] = s;
    }
}

// ---------------------------------------------------------------------------
// Host-side launcher
// ---------------------------------------------------------------------------
extern "C" void kernel_launch(void* const* d_in, const int* in_sizes, int n_in,
                              void* d_out, int out_size, void* d_ws, size_t ws_size,
                              hipStream_t stream)
{
    (void)in_sizes; (void)n_in; (void)out_size; (void)ws_size;

    const float* x        = (const float*)d_in[0];   // [N,128]
    const int*   edge     = (const int*)  d_in[1];   // [2,E]
    const float* conv1_W  = (const float*)d_in[2];   // [128,64]
    const float* conv1_as = (const float*)d_in[3];
    const float* conv1_ad = (const float*)d_in[4];
    const float* conv1_b  = (const float*)d_in[5];
    const float* convW    = (const float*)d_in[6];   // [4,64,64]
    const float* conv_as  = (const float*)d_in[7];   // [4,64]
    const float* conv_ad  = (const float*)d_in[8];
    const float* conv_b   = (const float*)d_in[9];
    const float* bn_g     = (const float*)d_in[10];  // [5,64]
    const float* bn_b     = (const float*)d_in[11];
    const float* bn_m     = (const float*)d_in[12];
    const float* bn_v     = (const float*)d_in[13];
    const float* projW    = (const float*)d_in[14];  // [4,64,64]
    const float* projb    = (const float*)d_in[15];
    const float* hW1      = (const float*)d_in[16];  // [64,32]
    const float* hb1      = (const float*)d_in[17];
    const float* hbn_g    = (const float*)d_in[18];
    const float* hbn_b    = (const float*)d_in[19];
    const float* hbn_m    = (const float*)d_in[20];
    const float* hbn_v    = (const float*)d_in[21];
    const float* hW2      = (const float*)d_in[22];  // [32,1]
    const float* hb2      = (const float*)d_in[23];

    const int* srcI = edge;        // edge_index[0]
    const int* dstI = edge + EE;   // edge_index[1]

    // Workspace partition (~40 MB of f32)
    float* ws     = (float*)d_ws;
    float* xp     = ws;                       // N*64
    float* hbuf   = xp   + (size_t)NN * 64;   // N*64 (h, later reused for proj)
    float* agg    = hbuf + (size_t)NN * 64;   // N*64
    float* asn    = agg  + (size_t)NN * 64;   // N
    float* adn    = asn  + NN;                // N
    float* emax   = adn  + NN;                // N
    float* denom  = emax + NN;                // N
    float* pooled = denom + NN;               // 64

    const dim3 blk256(256);
    const int  gM    = NN / 16;                    // 3125 WMMA row tiles
    const int  gNode = (NN + 255) / 256;
    const int  gNF   = (NN * 64 + 255) / 256;
    const int  gEdge = (EL + 255) / 256;
    const int  gEAgg = (EL * 32 + 255) / 256;      // wave per edge

    // ---------------- conv1 (K = 128) ----------------
    gemm_wmma_bf16<128><<<gM, 128, 0, stream>>>(x, conv1_W, hbuf, NN);
    attn_dots<<<gNode, blk256, 0, stream>>>(hbuf, conv1_as, conv1_ad, asn, adn, NN);
    init_layer<<<gNF, blk256, 0, stream>>>(agg, emax, denom, NN);
    edge_max<<<gEdge, blk256, 0, stream>>>(srcI, dstI, asn, adn, emax);
    edge_denom<<<gEdge, blk256, 0, stream>>>(srcI, dstI, asn, adn, emax, denom);
    edge_aggregate<<<gEAgg, blk256, 0, stream>>>(srcI, dstI, asn, adn, emax, denom, hbuf, agg);
    finish_layer<<<gNF, blk256, 0, stream>>>(agg, conv1_b,
        bn_g, bn_b, bn_m, bn_v, nullptr, nullptr, xp, /*useProj=*/0, NN);

    // ---------------- conv2..conv5 (K = 64) + residual proj ----------------
    for (int l = 0; l < 4; ++l) {
        const float* Wl  = convW + (size_t)l * 64 * 64;
        const float* Pl  = projW + (size_t)l * 64 * 64;
        gemm_wmma_bf16<64><<<gM, 128, 0, stream>>>(xp, Wl, hbuf, NN);
        attn_dots<<<gNode, blk256, 0, stream>>>(hbuf, conv_as + l * 64,
                                                conv_ad + l * 64, asn, adn, NN);
        init_layer<<<gNF, blk256, 0, stream>>>(agg, emax, denom, NN);
        edge_max<<<gEdge, blk256, 0, stream>>>(srcI, dstI, asn, adn, emax);
        edge_denom<<<gEdge, blk256, 0, stream>>>(srcI, dstI, asn, adn, emax, denom);
        edge_aggregate<<<gEAgg, blk256, 0, stream>>>(srcI, dstI, asn, adn, emax, denom, hbuf, agg);
        // residual projection: reuse hbuf (h no longer needed after aggregate)
        gemm_wmma_bf16<64><<<gM, 128, 0, stream>>>(xp, Pl, hbuf, NN);
        finish_layer<<<gNF, blk256, 0, stream>>>(agg, conv_b + l * 64,
            bn_g + (l + 1) * 64, bn_b + (l + 1) * 64,
            bn_m + (l + 1) * 64, bn_v + (l + 1) * 64,
            hbuf, projb + l * 64, xp, /*useProj=*/1, NN);
    }

    // ---------------- mean pool + MLP head ----------------
    init_pooled<<<1, 64, 0, stream>>>(pooled);
    pool_mean<<<(NN + 1023) / 1024, blk256, 0, stream>>>(xp, pooled, NN);
    head_kernel<<<1, 32, 0, stream>>>(pooled, hW1, hb1,
        hbn_g, hbn_b, hbn_m, hbn_v, hW2, hb2, (float*)d_out, NN);
}
